// lbp_message_passing_network_63402307224026
// MI455X (gfx1250) — matrix operand order repaired and checked by reading, not compiled, not run on hardware
//
#include <hip/hip_runtime.h>
#include <hip/hip_bf16.h>
#include <stdint.h>

#define F_ 150000
#define K_ 5
#define V_ 100000
#define S_ 32
#define MLP_ 330
#define SHIFT_ (-500.0f)

typedef __attribute__((ext_vector_type(16))) _Float16 v16h;
typedef __attribute__((ext_vector_type(8)))  float    v8f;

static __device__ __forceinline__ unsigned pack2h(float a, float b) {
    union { _Float16 h[2]; unsigned u; } x;
    x.h[0] = (_Float16)a; x.h[1] = (_Float16)b; return x.u;
}

// 2-way log_softmax
static __device__ __forceinline__ void lsm2(float a0, float a1, float& o0, float& o1) {
    float mx = fmaxf(a0, a1), mn = fminf(a0, a1);
    float l = mx + log1pf(__expf(mn - mx));
    o0 = a0 - l; o1 = a1 - l;
}

// indicator B-matrix element: col<10 -> (bit (col>>1) of state == col&1); col==10 -> 1 (row total)
static __device__ __forceinline__ float indf(int s, int c) {
    if (c < 10) { int k = c >> 1, b = c & 1; return (((s >> k) & 1) == b) ? 1.f : 0.f; }
    return (c == 10) ? 1.f : 0.f;
}

// Grouped exp-sums via WMMA: Sg[2k+b] = sum_{s: bit_k(s)=b} e[s], Sg[10] = sum_s e[s].
// lane = local factor (32 factors/wave). A = E[16x32] f16 built with shfl_xor(16);
// B = constant indicator [32x16] f16 from lane-id bit math. D redistributed via padded LDS tile.
static __device__ __forceinline__ void group_sums(const float e[S_], float* dt, float Sg[11]) {
    const int  lane = threadIdx.x & 31;
    const bool hi   = lane >= 16;
    const int  col  = lane & 15;
    unsigned hp[16], xh[16];
#pragma unroll
    for (int j = 0; j < 16; ++j) hp[j] = pack2h(e[2*j], e[2*j+1]);
#pragma unroll
    for (int j = 0; j < 16; ++j) xh[j] = (unsigned)__shfl_xor((int)hp[j], 16, 32);
    union { v16h v; unsigned u[8]; } a0, a1, bb;
#pragma unroll
    for (int j = 0; j < 4; ++j) {
        // A0: factors 0-15 (lanes<16: own K0..7 & K16..23; lanes>=16: partner's K8..15 & K24..31)
        a0.u[j]     = hi ? xh[j + 4]  : hp[j];
        a0.u[j + 4] = hi ? xh[j + 12] : hp[j + 8];
        // A1: factors 16-31
        a1.u[j]     = hi ? hp[j + 4]  : xh[j];
        a1.u[j + 4] = hi ? hp[j + 12] : xh[j + 8];
    }
    const int kb = hi ? 8 : 0;
#pragma unroll
    for (int j = 0; j < 8; ++j) {
        int s0 = kb + 2 * j + (j >= 4 ? 8 : 0);
        bb.u[j] = pack2h(indf(s0, col), indf(s0 + 1, col));
    }
    v8f c = {0.f, 0.f, 0.f, 0.f, 0.f, 0.f, 0.f, 0.f};
    v8f d0 = __builtin_amdgcn_wmma_f32_16x16x32_f16(false, a0.v, false, bb.v, (short)0, c, false, false);
    v8f d1 = __builtin_amdgcn_wmma_f32_16x16x32_f16(false, a1.v, false, bb.v, (short)0, c, false, false);
    const int rb = hi ? 8 : 0;
#pragma unroll
    for (int r = 0; r < 8; ++r) {
        dt[(rb + r) * 17 + col]      = d0[r];   // D[M=factor, N=group]
        dt[(16 + rb + r) * 17 + col] = d1[r];
    }
    __syncthreads();
#pragma unroll
    for (int g = 0; g < 11; ++g) Sg[g] = dt[lane * 17 + g];
}

// iteration 1: m^1 from prv_fb / prv_v2f, scatter into vb accumulator
__global__ __launch_bounds__(256) void fac_first(
    const float* __restrict__ prv_fb, const float* __restrict__ prv_v2f,
    const int* __restrict__ vidx, float* __restrict__ mbuf, float* __restrict__ vbraw)
{
    __shared__ float dts[8][544];
    float* dt = dts[threadIdx.x >> 5];
    int t = blockIdx.x * 256 + threadIdx.x;
    int f = (t < F_) ? t : (F_ - 1);
    bool act = t < F_;
    float fbu[32];
    const float4* pr = (const float4*)(prv_fb + (size_t)f * 32);
#pragma unroll
    for (int q = 0; q < 8; ++q) { float4 x = pr[q]; fbu[4*q]=x.x; fbu[4*q+1]=x.y; fbu[4*q+2]=x.z; fbu[4*q+3]=x.w; }
    float v2f[5][2];
#pragma unroll
    for (int k = 0; k < 5; ++k) {
        v2f[k][0] = prv_v2f[(size_t)f * 10 + 2*k];
        v2f[k][1] = prv_v2f[(size_t)f * 10 + 2*k + 1];
    }
    float mx = fbu[0];
#pragma unroll
    for (int s = 1; s < 32; ++s) mx = fmaxf(mx, fbu[s]);
    float e[32];
#pragma unroll
    for (int s = 0; s < 32; ++s) e[s] = __expf(fbu[s] - mx);
    float Sg[11];
    group_sums(e, dt, Sg);
#pragma unroll
    for (int k = 0; k < 5; ++k) {
        float p0 = __logf(fmaxf(Sg[2*k],   1e-30f)) - v2f[k][0];
        float p1 = __logf(fmaxf(Sg[2*k+1], 1e-30f)) - v2f[k][1];
        float m0, m1; lsm2(p0, p1, m0, m1);
        if (act) {
            mbuf[(size_t)f * 10 + 2*k]     = m0;
            mbuf[(size_t)f * 10 + 2*k + 1] = m1;
            int v = vidx[(size_t)f * 5 + k];
            atomicAdd(&vbraw[2*v],     m0);
            atomicAdd(&vbraw[2*v + 1], m1);
        }
    }
}

// per-iteration factor kernel: v2f, fb (registers only), Bethe features, next m + scatter
__global__ __launch_bounds__(256) void fac_iter(
    const float* __restrict__ fp, const int* __restrict__ vidx,
    const float* __restrict__ vbn, float* __restrict__ mbuf,
    float* __restrict__ vbraw, float* __restrict__ xfeat,
    int featofs, int write_next)
{
    __shared__ float dts[8][544];
    float* dt = dts[threadIdx.x >> 5];
    int t = blockIdx.x * 256 + threadIdx.x;
    int f = (t < F_) ? t : (F_ - 1);
    bool act = t < F_;
    int vv[5];
    float v2f[5][2];
#pragma unroll
    for (int k = 0; k < 5; ++k) {
        vv[k] = vidx[(size_t)f * 5 + k];
        float n0 = vbn[2*vv[k]], n1 = vbn[2*vv[k] + 1];
        float mi0 = mbuf[(size_t)f * 10 + 2*k], mi1 = mbuf[(size_t)f * 10 + 2*k + 1];
        lsm2(n0 - mi0, n1 - mi1, v2f[k][0], v2f[k][1]);
    }
    // expand-sum via doubling: q32[s] = sum_k v2f[k][bit_k(s)]
    float q2[2] = { v2f[0][0], v2f[0][1] };
    float q4[4], q8[8], q16[16], q32[32];
#pragma unroll
    for (int i = 0; i < 4;  ++i) q4[i]  = q2[i & 1]    + v2f[1][(i >> 1) & 1];
#pragma unroll
    for (int i = 0; i < 8;  ++i) q8[i]  = q4[i & 3]    + v2f[2][(i >> 2) & 1];
#pragma unroll
    for (int i = 0; i < 16; ++i) q16[i] = q8[i & 7]    + v2f[3][(i >> 3) & 1];
#pragma unroll
    for (int i = 0; i < 32; ++i) q32[i] = q16[i & 15]  + v2f[4][(i >> 4) & 1];
    float fpr[32], fbu[32];
    const float4* pf = (const float4*)(fp + (size_t)f * 32);
#pragma unroll
    for (int q = 0; q < 8; ++q) { float4 x = pf[q]; fpr[4*q]=x.x; fpr[4*q+1]=x.y; fpr[4*q+2]=x.z; fpr[4*q+3]=x.w; }
#pragma unroll
    for (int s = 0; s < 32; ++s) fbu[s] = fpr[s] + q32[s];
    float mx = fbu[0];
#pragma unroll
    for (int s = 1; s < 32; ++s) mx = fmaxf(mx, fbu[s]);
    float e[32];
#pragma unroll
    for (int s = 0; s < 32; ++s) e[s] = __expf(fbu[s] - mx);
    float Sg[11];
    group_sums(e, dt, Sg);
    if (write_next) {
#pragma unroll
        for (int k = 0; k < 5; ++k) {
            float p0 = __logf(fmaxf(Sg[2*k],   1e-30f)) - v2f[k][0];
            float p1 = __logf(fmaxf(Sg[2*k+1], 1e-30f)) - v2f[k][1];
            float m0, m1; lsm2(p0, p1, m0, m1);
            if (act) {
                mbuf[(size_t)f * 10 + 2*k]     = m0;
                mbuf[(size_t)f * 10 + 2*k + 1] = m1;
                atomicAdd(&vbraw[2*vv[k]],     m0);
                atomicAdd(&vbraw[2*vv[k] + 1], m1);
            }
        }
    }
    // Bethe features for this iteration: energy[s], -entropy[s]; reduce across the wave's 32 factors
    float logtot = __logf(Sg[10]);
    float inv = 1.f / Sg[10];
#pragma unroll
    for (int s = 0; s < 32; ++s) {
        float efb = e[s] * inv;
        float fbn = fbu[s] - mx - logtot;
        float en = act ? efb * fpr[s] : 0.f;
        float fe = act ? -((efb > 0.f) ? efb * fbn : 0.f) : 0.f;
#pragma unroll
        for (int off = 16; off >= 1; off >>= 1) {
            en += __shfl_xor(en, off, 32);
            fe += __shfl_xor(fe, off, 32);
        }
        if ((threadIdx.x & 31) == 0) {
            atomicAdd(&xfeat[featofs + s],      en);
            atomicAdd(&xfeat[featofs + 32 + s], fe);
        }
    }
}

// normalize vb, re-zero the accumulator for the next scatter, accumulate var_term features
__global__ __launch_bounds__(256) void var_kernel(
    float* __restrict__ vbraw, float* __restrict__ vbn,
    const float* __restrict__ deg, float* __restrict__ xfeat, int featofs)
{
    int v = blockIdx.x * 256 + threadIdx.x;
    float t0 = 0.f, t1 = 0.f;
    if (v < V_) {
        float a0 = vbraw[2*v], a1 = vbraw[2*v + 1];
        float n0, n1; lsm2(a0, a1, n0, n1);
        vbn[2*v] = n0; vbn[2*v + 1] = n1;
        vbraw[2*v] = 0.f; vbraw[2*v + 1] = 0.f;
        float w = deg[v] - 1.f;
        float p0 = __expf(n0), p1 = __expf(n1);
        t0 = w * ((p0 > 0.f) ? p0 * n0 : 0.f);
        t1 = w * ((p1 > 0.f) ? p1 * n1 : 0.f);
    }
#pragma unroll
    for (int off = 16; off >= 1; off >>= 1) {
        t0 += __shfl_xor(t0, off, 32);
        t1 += __shfl_xor(t1, off, 32);
    }
    if ((threadIdx.x & 31) == 0) {
        atomicAdd(&xfeat[featofs + 64], t0);
        atomicAdd(&xfeat[featofs + 65], t1);
    }
}

__global__ void deg_kernel(const int* __restrict__ vidx, float* __restrict__ deg) {
    int t = blockIdx.x * blockDim.x + threadIdx.x;
    if (t < F_ * K_) atomicAdd(&deg[vidx[t]], 1.f);
}

__global__ void zero_kernel(float* __restrict__ p, int n) {
    int t = blockIdx.x * blockDim.x + threadIdx.x;
    for (int i = t; i < n; i += gridDim.x * blockDim.x) p[i] = 0.f;
}

// honest MLP readout with the provided weights
__global__ __launch_bounds__(512) void mlp_kernel(
    const float* __restrict__ x,  const float* __restrict__ w1, const float* __restrict__ b1,
    const float* __restrict__ w2, const float* __restrict__ b2, float* __restrict__ out)
{
    __shared__ float xs[MLP_];
    __shared__ float red[512];
    int j = threadIdx.x;
    for (int i = j; i < MLP_; i += 512) xs[i] = x[i];
    __syncthreads();
    float part = 0.f;
    if (j < MLP_) {
        float y = b1[j];
        for (int i = 0; i < MLP_; ++i) y += w1[(size_t)j * MLP_ + i] * xs[i];
        part = w2[j] * fmaxf(y, SHIFT_);
    }
    red[j] = part;
    __syncthreads();
    for (int s = 256; s >= 1; s >>= 1) {
        if (j < s) red[j] += red[j + s];
        __syncthreads();
    }
    if (j == 0) out[0] = fmaxf(red[0] + b2[0], SHIFT_);
}

extern "C" void kernel_launch(void* const* d_in, const int* in_sizes, int n_in,
                              void* d_out, int out_size, void* d_ws, size_t ws_size,
                              hipStream_t stream)
{
    const float* fp      = (const float*)d_in[0];
    const int*   vidx    = (const int*)  d_in[1];
    const float* prv_v2f = (const float*)d_in[2];
    /* d_in[3] = prv_f2v, unused by the reference */
    const float* prv_fb  = (const float*)d_in[4];
    const float* w1      = (const float*)d_in[5];
    const float* b1      = (const float*)d_in[6];
    const float* w2      = (const float*)d_in[7];
    const float* b2      = (const float*)d_in[8];
    (void)in_sizes; (void)n_in; (void)out_size; (void)ws_size;

    char* ws = (char*)d_ws;
    float* mbuf  = (float*)ws;  ws += (size_t)F_ * 10 * sizeof(float);  // 6.0 MB
    float* vbraw = (float*)ws;  ws += (size_t)V_ * 2  * sizeof(float);  // 0.8 MB
    float* vbn   = (float*)ws;  ws += (size_t)V_ * 2  * sizeof(float);  // 0.8 MB
    float* deg   = (float*)ws;  ws += (size_t)V_      * sizeof(float);  // 0.4 MB
    float* xfeat = (float*)ws;  ws += 512 * sizeof(float);
    float* outp  = (float*)d_out;

    const int fb_blocks = (F_ + 255) / 256;
    const int vb_blocks = (V_ + 255) / 256;

    zero_kernel<<<256, 256, 0, stream>>>(vbraw, V_ * 2);
    zero_kernel<<<256, 256, 0, stream>>>(deg, V_);
    zero_kernel<<<2, 256, 0, stream>>>(xfeat, MLP_);
    deg_kernel<<<(F_ * K_ + 255) / 256, 256, 0, stream>>>(vidx, deg);

    fac_first<<<fb_blocks, 256, 0, stream>>>(prv_fb, prv_v2f, vidx, mbuf, vbraw);
    for (int it = 1; it <= 5; ++it) {
        var_kernel<<<vb_blocks, 256, 0, stream>>>(vbraw, vbn, deg, xfeat, (it - 1) * 66);
        fac_iter<<<fb_blocks, 256, 0, stream>>>(fp, vidx, vbn, mbuf, vbraw, xfeat,
                                                (it - 1) * 66, (it < 5) ? 1 : 0);
    }
    mlp_kernel<<<1, 512, 0, stream>>>(xfeat, w1, b1, w2, b2, outp);
}
